// SLAYTensorAttention_5377299054795
// MI455X (gfx1250) — matrix operand order repaired and checked by reading, not compiled, not run on hardware
//
#include <hip/hip_runtime.h>
#include <hip/hip_bf16.h>

// ---------------------------------------------------------------------------
// Types for WMMA fragments (CDNA5 / gfx1250, wave32)
// ---------------------------------------------------------------------------
typedef __attribute__((ext_vector_type(16))) _Float16 v16h;
typedef __attribute__((ext_vector_type(8)))  _Float16 v8h;
typedef __attribute__((ext_vector_type(8)))  float    v8f;

union FragU { v8h h2[2]; v16h v; };

// ---------------------------------------------------------------------------
// Problem constants
// ---------------------------------------------------------------------------
#define BB 4
#define LL 4096
#define EE 1024
#define HH 16
#define DHH 64
#define PP 16
#define MM 8
#define FF 128      // R*P*M
#define KSPLIT 8    // split-K factor for the kv GEMM

// ---------------------------------------------------------------------------
// f32 -> f16 conversion (n multiple of 8)
// ---------------------------------------------------------------------------
__global__ __launch_bounds__(256) void cvt_kernel(
    const float* __restrict__ src, _Float16* __restrict__ dst, long long n)
{
    const long long i = ((long long)blockIdx.x * 256 + threadIdx.x) * 8;
    if (i >= n) return;
#pragma unroll
    for (int j = 0; j < 8; ++j) dst[i + j] = (_Float16)src[i + j];
}

// ---------------------------------------------------------------------------
// Large dense GEMM:  C(M,N) = A(M,K)f16 * B(K,N)f16 [+ bias]
//   256 threads = 8 waves; macro-tile 128x128; K step 32.
//   Wave w: M-group (w&3)*32 (2 strips of 16), N-group (w>>2)*64 (4 subtiles)
//   -> 8 WMMAs / wave / K-step, 64 WMMAs (1 MFLOP) per 16KB staged.
// Fragment layouts per CDNA5 ISA §7.12.2 (same as round 1).
// Requires M%128==0, N%128==0, K%32==0, A/B 16B-aligned, lda==K row-major.
// ---------------------------------------------------------------------------
template <typename CT>
__global__ __launch_bounds__(256) void gemm_wmma128(
    const _Float16* __restrict__ A, const _Float16* __restrict__ Bm,
    const float* __restrict__ bias, CT* __restrict__ C,
    int M, int N, int K)
{
    __shared__ alignas(16) _Float16 As[128 * 40]; // [row m][k], stride 40
    __shared__ alignas(16) _Float16 Bs[128 * 40]; // [col n][k], stride 40

    const int tid  = threadIdx.x;
    const int wave = tid >> 5;
    const int lane = tid & 31;
    const int n0 = blockIdx.x * 128;
    const int m0 = blockIdx.y * 128;
    const int wm = (wave & 3) * 32;   // wave M offset
    const int wn = (wave >> 2) * 64;  // wave N offset

    v8f acc[2][4] = {};

    const int mrow   = lane & 15;
    const int khalfA = (lane < 16) ? 0 : 8;
    const int khalfB = (lane < 16) ? 0 : 8;

    const int rA  = tid >> 1;            // A stage: row (0..127)
    const int hA  = (tid & 1) * 16;      //          16-half group
    const int kB  = tid & 31;            // B stage: k row
    const int nbB = (tid >> 5) * 16;     //          16-col group (0..112)

    for (int k0 = 0; k0 < K; k0 += 32) {
        // ---- stage A tile (128x32) f16 -> LDS (pure vector copy) ----
        {
            const _Float16* src = A + (long long)(m0 + rA) * K + (k0 + hA);
            _Float16* dst = &As[rA * 40 + hA];
            if (k0 + 32 < K) __builtin_prefetch(src + 32, 0, 1);
            *(v8h*)dst       = *(const v8h*)src;
            *(v8h*)(dst + 8) = *(const v8h*)(src + 8);
        }
        // ---- stage B tile (32x128) transposed into LDS ----
        {
            const _Float16* src = Bm + (long long)(k0 + kB) * N + (n0 + nbB);
            if (k0 + 32 < K) __builtin_prefetch(src + (long long)32 * N, 0, 1);
#pragma unroll
            for (int j = 0; j < 16; ++j)
                Bs[(nbB + j) * 40 + kB] = src[j];
        }
        __syncthreads();

        FragU af[2];
#pragma unroll
        for (int mt = 0; mt < 2; ++mt) {
            const _Float16* ap = &As[(wm + mt * 16 + mrow) * 40 + khalfA];
            af[mt].h2[0] = *(const v8h*)ap;
            af[mt].h2[1] = *(const v8h*)(ap + 16);
        }
#pragma unroll
        for (int nt = 0; nt < 4; ++nt) {
            FragU bf;
            const _Float16* bp = &Bs[(wn + nt * 16 + mrow) * 40 + khalfB];
            bf.h2[0] = *(const v8h*)bp;
            bf.h2[1] = *(const v8h*)(bp + 16);
#pragma unroll
            for (int mt = 0; mt < 2; ++mt)
                acc[mt][nt] = __builtin_amdgcn_wmma_f32_16x16x32_f16(
                    false, af[mt].v, false, bf.v, (short)0, acc[mt][nt], false, false);
        }
        __syncthreads();
    }

    const int rowSel = (lane >> 4) * 8;
#pragma unroll
    for (int nt = 0; nt < 4; ++nt) {
        const int col = n0 + wn + nt * 16 + (lane & 15);
        const float bv = bias ? bias[col] : 0.0f;
#pragma unroll
        for (int mt = 0; mt < 2; ++mt)
#pragma unroll
            for (int r = 0; r < 8; ++r) {
                const int row = m0 + wm + mt * 16 + rowSel + r;
                C[(long long)row * N + col] = (CT)(acc[mt][nt][r] + bv);
            }
    }
}

// ---------------------------------------------------------------------------
// Small batched GEMM (64x64 tile, 4 waves) with explicit leading dims and a
// two-level batch decomposition: blockIdx.z = z1*zdiv + z2 (z2 = split-K id).
// A/B converted to f16 while staging if needed.
// ---------------------------------------------------------------------------
template <typename AT, typename BT, typename CT>
__global__ __launch_bounds__(128) void gemm_wmma64(
    const AT* __restrict__ A, const BT* __restrict__ Bm,
    const float* __restrict__ bias, CT* __restrict__ C,
    int M, int N, int K, int lda, int ldb, int ldc, int zdiv,
    long long bsA1, long long bsA2,
    long long bsB1, long long bsB2,
    long long bsC1, long long bsC2)
{
    __shared__ alignas(16) _Float16 As[64 * 40];
    __shared__ alignas(16) _Float16 Bs[64 * 40];

    const int tid  = threadIdx.x;
    const int wave = tid >> 5;
    const int lane = tid & 31;
    const int n0 = blockIdx.x * 64;
    const int m0 = blockIdx.y * 64;

    const int z1 = blockIdx.z / zdiv;
    const int z2 = blockIdx.z % zdiv;
    A  += (long long)z1 * bsA1 + (long long)z2 * bsA2;
    Bm += (long long)z1 * bsB1 + (long long)z2 * bsB2;
    C  += (long long)z1 * bsC1 + (long long)z2 * bsC2;

    v8f acc[4] = {};

    const int mrow   = lane & 15;
    const int khalfA = (lane < 16) ? 0 : 8;
    const int khalfB = (lane < 16) ? 0 : 8;

    const int rA  = tid >> 1;
    const int hA  = (tid & 1) * 16;
    const int kB  = tid & 31;
    const int nbB = (tid >> 5) * 16;

    for (int k0 = 0; k0 < K; k0 += 32) {
        {
            const AT* src = A + (long long)(m0 + rA) * lda + (k0 + hA);
            _Float16* dst = &As[rA * 40 + hA];
            if (k0 + 32 < K) __builtin_prefetch(src + 32, 0, 1);
#pragma unroll
            for (int j = 0; j < 16; ++j) dst[j] = (_Float16)(float)src[j];
        }
        {
            const BT* src = Bm + (long long)(k0 + kB) * ldb + (n0 + nbB);
            if (k0 + 32 < K) __builtin_prefetch(src + (long long)32 * ldb, 0, 1);
#pragma unroll
            for (int j = 0; j < 16; ++j)
                Bs[(nbB + j) * 40 + kB] = (_Float16)(float)src[j];
        }
        __syncthreads();

        FragU af;
        {
            const _Float16* ap = &As[(wave * 16 + mrow) * 40 + khalfA];
            af.h2[0] = *(const v8h*)ap;
            af.h2[1] = *(const v8h*)(ap + 16);
        }
#pragma unroll
        for (int nt = 0; nt < 4; ++nt) {
            FragU bf;
            const _Float16* bp = &Bs[(nt * 16 + mrow) * 40 + khalfB];
            bf.h2[0] = *(const v8h*)bp;
            bf.h2[1] = *(const v8h*)(bp + 16);
            acc[nt] = __builtin_amdgcn_wmma_f32_16x16x32_f16(
                false, af.v, false, bf.v, (short)0, acc[nt], false, false);
        }
        __syncthreads();
    }

    const int rowSel = (lane >> 4) * 8;
#pragma unroll
    for (int nt = 0; nt < 4; ++nt) {
        const int col = n0 + nt * 16 + (lane & 15);
        const float bv = bias ? bias[col] : 0.0f;
#pragma unroll
        for (int r = 0; r < 8; ++r) {
            const int row = m0 + wave * 16 + rowSel + r;
            C[(long long)row * ldc + col] = (CT)(acc[nt][r] + bv);
        }
    }
}

// ---------------------------------------------------------------------------
// Feature kernel (unchanged from round 1): per token/head
//   L2-normalize -> count-sketch x2 -> circular conv (== irfft(rfft*rfft))
//   -> PRF exp features -> outer-product fuse (F=128)
// ---------------------------------------------------------------------------
__global__ __launch_bounds__(64) void feature_kernel(
    const _Float16* __restrict__ qkv,
    const float* __restrict__ omega,
    const int* __restrict__ h1, const int* __restrict__ s1,
    const int* __restrict__ h2, const int* __restrict__ s2,
    const float* __restrict__ qnodes, const float* __restrict__ qweights,
    _Float16* __restrict__ q_fuse,
    _Float16* __restrict__ k_fuseT,
    _Float16* __restrict__ v_h)
{
    const int bh = blockIdx.y;
    const int b  = bh >> 4;
    const int h  = bh & 15;
    const int l0 = blockIdx.x * 64;
    const int tid = threadIdx.x;

    __shared__ float qs[64][65];
    __shared__ float ks[64][65];
    __shared__ float om[64][8];
    __shared__ int   hh1[64], hh2[64];
    __shared__ float s1f[64], s2f[64];

    hh1[tid] = h1[tid]; s1f[tid] = (float)s1[tid];
    hh2[tid] = h2[tid]; s2f[tid] = (float)s2[tid];
    for (int i = tid; i < 64 * 8; i += 64) om[i >> 3][i & 7] = omega[h * 512 + i];

    for (int i = tid; i < 64 * 64; i += 64) {
        const int t = i >> 6, d = i & 63;
        const long long base = ((long long)(b * LL + l0 + t)) * (3 * EE) + h * DHH + d;
        qs[t][d] = (float)qkv[base];
        ks[t][d] = (float)qkv[base + EE];
        v_h[((long long)bh * LL + l0 + t) * DHH + d] = qkv[base + 2 * EE];
    }
    __syncthreads();

    const float s_node = qnodes[0];
    const float sq2s   = sqrtf(2.0f * fmaxf(s_node, 0.0f));
    const float sw     = sqrtf(fmaxf(qweights[0], 0.0f));
    const int t = tid;
    const int l = l0 + t;

    for (int which = 0; which < 2; ++which) {
        float (*xs)[65] = (which == 0) ? qs : ks;

        float nrm = 0.0f;
        for (int d = 0; d < 64; ++d) { const float xv = xs[t][d]; nrm += xv * xv; }
        const float inv = 1.0f / fmaxf(sqrtf(nrm), 1e-6f);
        for (int d = 0; d < 64; ++d) xs[t][d] *= inv;

        float cs1[16], cs2[16];
#pragma unroll
        for (int p = 0; p < 16; ++p) {
            float c1 = 0.0f, c2 = 0.0f;
            for (int d = 0; d < 64; ++d) {
                const float xv = xs[t][d];
                c1 += (hh1[d] == p) ? s1f[d] * xv : 0.0f;
                c2 += (hh2[d] == p) ? s2f[d] * xv : 0.0f;
            }
            cs1[p] = c1; cs2[p] = c2;
        }

        float proj[8] = {};
        for (int d = 0; d < 64; ++d) {
            const float xv = xs[t][d];
#pragma unroll
            for (int m = 0; m < 8; ++m) proj[m] += xv * om[d][m];
        }

        float poly[16];
#pragma unroll
        for (int p = 0; p < 16; ++p) {
            float a = 0.0f;
#pragma unroll
            for (int j = 0; j < 16; ++j) a += cs1[j] * cs2[(p - j) & 15];
            poly[p] = a * 0.25f; // 1/sqrt(P)
        }

        float prf[8];
#pragma unroll
        for (int m = 0; m < 8; ++m) {
            float e = proj[m] * sq2s - s_node;
            e = fminf(fmaxf(e, -10.0f), 10.0f);
            prf[m] = expf(e) * 0.3535533905932738f * sw; // 1/sqrt(M)*sqrt(qw)
        }

        if (which == 0) {
            _Float16* dst = q_fuse + (((long long)bh * LL) + l) * FF;
#pragma unroll
            for (int p = 0; p < 16; ++p)
#pragma unroll
                for (int m = 0; m < 8; ++m)
                    dst[p * 8 + m] = (_Float16)(poly[p] * prf[m]);
        } else {
            const long long base = ((long long)bh * FF) * LL + l;
#pragma unroll
            for (int p = 0; p < 16; ++p)
#pragma unroll
                for (int m = 0; m < 8; ++m)
                    k_fuseT[base + (long long)(p * 8 + m) * LL] =
                        (_Float16)(poly[p] * prf[m]);
        }
    }
}

// ---------------------------------------------------------------------------
// k_sum[bh][f] = sum_l k_fuseT[bh][f][l]
// ---------------------------------------------------------------------------
__global__ __launch_bounds__(256) void ksum_kernel(
    const _Float16* __restrict__ kT, float* __restrict__ ksum)
{
    const long long row = blockIdx.x;
    const _Float16* src = kT + row * LL;
    float s = 0.0f;
    for (int i = threadIdx.x; i < LL; i += 256) s += (float)src[i];
    __shared__ float red[256];
    red[threadIdx.x] = s;
    __syncthreads();
    for (int off = 128; off > 0; off >>= 1) {
        if (threadIdx.x < off) red[threadIdx.x] += red[threadIdx.x + off];
        __syncthreads();
    }
    if (threadIdx.x == 0) ksum[row] = red[0];
}

// ---------------------------------------------------------------------------
// kv split-K reduction: kv[bh][f][d] = sum_s part[bh][s][f][d]
// ---------------------------------------------------------------------------
__global__ __launch_bounds__(256) void kv_reduce_kernel(
    const float* __restrict__ part, float* __restrict__ kv)
{
    const long long i = (long long)blockIdx.x * 256 + threadIdx.x; // bh*8192+f*64+d
    const long long bh  = i >> 13;
    const long long off = i & 8191;
    const float* p = part + bh * (KSPLIT * FF * DHH) + off;
    float s = 0.0f;
#pragma unroll
    for (int sp = 0; sp < KSPLIT; ++sp) s += p[(long long)sp * FF * DHH];
    kv[i] = s;
}

// ---------------------------------------------------------------------------
// norm = q_fuse . k_sum ; attn(B,L,E) f16 = context / max(norm, 1e-4)
// ---------------------------------------------------------------------------
__global__ __launch_bounds__(256) void norm_kernel(
    const float* __restrict__ context,
    const _Float16* __restrict__ q_fuse,
    const float* __restrict__ ksum,
    _Float16* __restrict__ attn)
{
    const long long idx = (long long)blockIdx.x * 256 + threadIdx.x; // bh*L + l
    const long long bh  = idx >> 12;
    const long long l   = idx & (LL - 1);
    const int b = (int)(bh >> 4), h = (int)(bh & 15);

    const _Float16* qf = q_fuse + idx * FF;
    const float*    ks = ksum + bh * FF;
    float nrm = 0.0f;
    for (int f = 0; f < FF; ++f) nrm += (float)qf[f] * ks[f];
    const float inv = 1.0f / fmaxf(nrm, 1e-4f);

    const float* ctx = context + idx * DHH;
    _Float16* dst = attn + ((long long)b * LL + l) * EE + h * DHH;
#pragma unroll
    for (int d = 0; d < DHH; ++d) dst[d] = (_Float16)(ctx[d] * inv);
}

// ---------------------------------------------------------------------------
// Launch
// ---------------------------------------------------------------------------
extern "C" void kernel_launch(void* const* d_in, const int* in_sizes, int n_in,
                              void* d_out, int out_size, void* d_ws, size_t ws_size,
                              hipStream_t stream)
{
    (void)in_sizes; (void)n_in; (void)out_size; (void)ws_size;

    const float* x      = (const float*)d_in[0];
    const float* qkv_w  = (const float*)d_in[1];
    const float* qkv_b  = (const float*)d_in[2];
    const float* out_w  = (const float*)d_in[3];
    const float* out_b  = (const float*)d_in[4];
    const float* omega  = (const float*)d_in[5];
    const float* qnodes = (const float*)d_in[6];
    const float* qwts   = (const float*)d_in[7];
    const int*   tsh1   = (const int*)d_in[8];
    const int*   tss1   = (const int*)d_in[9];
    const int*   tsh2   = (const int*)d_in[10];
    const int*   tss2   = (const int*)d_in[11];

    // workspace carve-out
    char* ws = (char*)d_ws;
    constexpr long long N_X   = (long long)BB * LL * EE;          // 16.7M
    constexpr long long N_QW  = (long long)EE * 3 * EE;           // 3.1M
    constexpr long long N_OW  = (long long)EE * EE;               // 1.0M
    constexpr long long SZ_XH   = N_X * 2;
    constexpr long long SZ_QWH  = N_QW * 2;
    constexpr long long SZ_OWH  = N_OW * 2;
    constexpr long long SZ_QKV  = (long long)BB * LL * 3 * EE * 2;
    constexpr long long SZ_VH   = (long long)BB * HH * LL * DHH * 2;
    constexpr long long SZ_QF   = (long long)BB * HH * LL * FF * 2;
    constexpr long long SZ_KFT  = (long long)BB * HH * FF * LL * 2;
    constexpr long long SZ_KVP  = (long long)BB * HH * KSPLIT * FF * DHH * 4;
    constexpr long long SZ_KV   = (long long)BB * HH * FF * DHH * 4;
    constexpr long long SZ_KSUM = (long long)BB * HH * FF * 4;
    constexpr long long SZ_CTX  = (long long)BB * HH * LL * DHH * 4;

    _Float16* x_h    = (_Float16*)ws;  ws += SZ_XH;
    _Float16* qw_h   = (_Float16*)ws;  ws += SZ_QWH;
    _Float16* ow_h   = (_Float16*)ws;  ws += SZ_OWH;
    _Float16* qkv_h  = (_Float16*)ws;  ws += SZ_QKV;
    _Float16* v_h    = (_Float16*)ws;  ws += SZ_VH;
    _Float16* q_fuse = (_Float16*)ws;  ws += SZ_QF;
    _Float16* k_fT   = (_Float16*)ws;  ws += SZ_KFT;
    float*    kv_p   = (float*)ws;     ws += SZ_KVP;
    float*    kv     = (float*)ws;     ws += SZ_KV;
    float*    ksum   = (float*)ws;     ws += SZ_KSUM;
    float*    ctx    = (float*)ws;     ws += SZ_CTX;
    _Float16* attn_h = (_Float16*)ws;  /* BB*LL*EE*2 */

    // 0) one-time f32 -> f16 conversions (removes cvt from GEMM hot loops)
    cvt_kernel<<<(unsigned)(N_X  / 8 / 256), 256, 0, stream>>>(x,     x_h,  N_X);
    cvt_kernel<<<(unsigned)(N_QW / 8 / 256), 256, 0, stream>>>(qkv_w, qw_h, N_QW);
    cvt_kernel<<<(unsigned)(N_OW / 8 / 256), 256, 0, stream>>>(out_w, ow_h, N_OW);

    // 1) QKV projection (16384x3072x1024), 128x128 macro-tile
    gemm_wmma128<_Float16>
        <<<dim3(3 * EE / 128, BB * LL / 128, 1), 256, 0, stream>>>(
            x_h, qw_h, qkv_b, qkv_h, BB * LL, 3 * EE, EE);

    // 2) feature construction
    feature_kernel<<<dim3(LL / 64, BB * HH, 1), 64, 0, stream>>>(
        qkv_h, omega, tsh1, tss1, tsh2, tss2, qnodes, qwts,
        q_fuse, k_fT, v_h);

    // 3) k_sum over L
    ksum_kernel<<<BB * HH * FF, 256, 0, stream>>>(k_fT, ksum);

    // 4) kv = k_fuseT(128x4096) * v(4096x64), split-K x8, batched over 64 bh
    gemm_wmma64<_Float16, _Float16, float>
        <<<dim3(1, FF / 64, BB * HH * KSPLIT), 128, 0, stream>>>(
            k_fT, v_h, nullptr, kv_p,
            FF, DHH, LL / KSPLIT, /*lda*/ LL, /*ldb*/ DHH, /*ldc*/ DHH,
            /*zdiv*/ KSPLIT,
            (long long)FF * LL,  (long long)(LL / KSPLIT),
            (long long)LL * DHH, (long long)(LL / KSPLIT) * DHH,
            (long long)KSPLIT * FF * DHH, (long long)FF * DHH);

    // 4b) reduce split-K partials
    kv_reduce_kernel<<<(BB * HH * FF * DHH) / 256, 256, 0, stream>>>(kv_p, kv);

    // 5) context = q_fuse(4096x128) * kv(128x64), batched over 64 bh
    gemm_wmma64<_Float16, float, float>
        <<<dim3(1, LL / 64, BB * HH), 128, 0, stream>>>(
            q_fuse, kv, nullptr, ctx,
            LL, DHH, FF, /*lda*/ FF, /*ldb*/ DHH, /*ldc*/ DHH,
            /*zdiv*/ 1,
            (long long)LL * FF, 0,
            (long long)FF * DHH, 0,
            (long long)LL * DHH, 0);

    // 6) normalize + repack to (B,L,E) f16
    norm_kernel<<<(BB * HH * LL) / 256, 256, 0, stream>>>(ctx, q_fuse, ksum, attn_h);

    // 7) output projection (16384x1024x1024) -> d_out f32
    gemm_wmma128<float>
        <<<dim3(EE / 128, BB * LL / 128, 1), 256, 0, stream>>>(
            attn_h, ow_h, out_b, (float*)d_out, BB * LL, EE, EE);
}
